// MultiLabelDiceLoss_31129922961540
// MI455X (gfx1250) — compile-verified, matrix operand order checked by reading, not run on hardware
//
#include <hip/hip_runtime.h>

typedef int v8i __attribute__((ext_vector_type(8)));

#define NUM_CLASSES 20

// Exact per-byte equality detect: returns 0x01 in each byte of v that equals the
// (replicated) byte in crep, 0x00 elsewhere. Works for all byte values.
__device__ __forceinline__ unsigned int onehot_eq(unsigned int v, unsigned int crep) {
    unsigned int x = v ^ crep;
    unsigned int t = ((x & 0x7F7F7F7Fu) + 0x7F7F7F7Fu) | x; // bit7 set iff byte nonzero
    return (~t & 0x80808080u) >> 7;                          // 0x01 at zero (==) bytes
}

__device__ __forceinline__ unsigned int bytesum(unsigned int x) {
    return (x & 0xFFu) + ((x >> 8) & 0xFFu) + ((x >> 16) & 0xFFu) + (x >> 24);
}

__global__ void dice_zero_bins(unsigned int* bins) {
    if (threadIdx.x < 64) bins[threadIdx.x] = 0u;
}

// bins layout (u32): [0..19] = hist(preds), [20..39] = hist(labels), [40..59] = hist(match)
__global__ __launch_bounds__(256) void dice_hist(const float* __restrict__ preds,
                                                 const float* __restrict__ labels,
                                                 unsigned int* __restrict__ bins,
                                                 int n) {
    __shared__ unsigned int smem[64];
    if (threadIdx.x < 64) smem[threadIdx.x] = 0u;
    __syncthreads();

    const int lane = (int)(threadIdx.x & 31u);
    const int wave = (int)(threadIdx.x >> 5);
    const int m    = lane & 15;   // WMMA A-matrix row = class id (0..15)
    const int h    = lane >> 4;   // lane half selects K-group per ISA A layout
    const unsigned int mrep = (unsigned int)m * 0x01010101u;

    // WMMA i32 accumulators: row m = count for class m (all 16 columns equal, B = ones)
    v8i accP = {}; v8i accL = {}; v8i accM = {};

    // B operand: 64x16 u8 all-ones (layout-independent)
    v8i bones;
#pragma unroll
    for (int i = 0; i < 8; ++i) bones[i] = 0x01010101;

    // A-operand source lanes: VGPR j of lane(h,m) needs packed u32 covering
    // K = 16*(j>>1) + 4*(j&1) + 8*h  (+64*b), i.e. source lane (K/4) of the chunk.
    int srcLane[8];
#pragma unroll
    for (int j = 0; j < 8; ++j) srcLane[j] = 4 * (j >> 1) + (j & 1) + 2 * h;

    // classes 16..19: per-lane packed byte accumulators (iters per wave is tiny)
    unsigned int hiP[4] = {0, 0, 0, 0}, hiL[4] = {0, 0, 0, 0}, hiM[4] = {0, 0, 0, 0};

    const long long cstride = (long long)gridDim.x * (long long)(blockDim.x >> 5) * 128ll;
    long long chunk = ((long long)blockIdx.x * (long long)(blockDim.x >> 5) + wave) * 128ll;

    for (; chunk < (long long)n; chunk += cstride) {   // wave-uniform: EXEC stays all-ones
        const long long base = chunk + (long long)(lane * 4);
        unsigned int pk = 0xFFFFFFFFu, lk = 0xFFFFFFFFu; // 0xFF = matches no class
        if (base + 4 <= (long long)n) {
            const float4 pf = *(const float4*)(preds + base);
            const float4 lf = *(const float4*)(labels + base);
            pk = ((unsigned int)pf.x) | (((unsigned int)pf.y) << 8) |
                 (((unsigned int)pf.z) << 16) | (((unsigned int)pf.w) << 24);
            lk = ((unsigned int)lf.x) | (((unsigned int)lf.y) << 8) |
                 (((unsigned int)lf.z) << 16) | (((unsigned int)lf.w) << 24);
        } else {
#pragma unroll
            for (int e = 0; e < 4; ++e) {
                if (base + e < (long long)n) {
                    unsigned int pb = ((unsigned int)preds[base + e]) & 0xFFu;
                    unsigned int lb = ((unsigned int)labels[base + e]) & 0xFFu;
                    pk = (pk & ~(0xFFu << (8 * e))) | (pb << (8 * e));
                    lk = (lk & ~(0xFFu << (8 * e))) | (lb << (8 * e));
                }
            }
        }

        // classes 0..15: one-hot A (16x64 u8) x ones (64x16 u8) on the matrix pipe
#pragma unroll
        for (int b = 0; b < 2; ++b) {
            v8i aP, aL, aM;
#pragma unroll
            for (int j = 0; j < 8; ++j) {
                unsigned int vp = (unsigned int)__shfl((int)pk, srcLane[j] + 16 * b);
                unsigned int vl = (unsigned int)__shfl((int)lk, srcLane[j] + 16 * b);
                unsigned int op = onehot_eq(vp, mrep);
                unsigned int ol = onehot_eq(vl, mrep);
                aP[j] = (int)op;
                aL[j] = (int)ol;
                aM[j] = (int)(op & ol);
            }
            accP = __builtin_amdgcn_wmma_i32_16x16x64_iu8(false, aP, false, bones, accP, false, false);
            accL = __builtin_amdgcn_wmma_i32_16x16x64_iu8(false, aL, false, bones, accL, false, false);
            accM = __builtin_amdgcn_wmma_i32_16x16x64_iu8(false, aM, false, bones, accM, false, false);
        }

        // classes 16..19: packed SWAR on each lane's own 4 elements
#pragma unroll
        for (int c = 0; c < 4; ++c) {
            const unsigned int crep = (unsigned int)(16 + c) * 0x01010101u;
            unsigned int zp = onehot_eq(pk, crep);
            unsigned int zl = onehot_eq(lk, crep);
            hiP[c] += zp;
            hiL[c] += zl;
            hiM[c] += (zp & zl);
        }
    }

    // ---- per-wave -> per-block reduction ----
    // D layout (16x16 i32): VGPR r holds M=r for lanes 0-15, M=r+8 for lanes 16-31;
    // all columns equal since B = ones. Lane 0 owns bins 0..7, lane 16 owns bins 8..15.
    if (lane == 0) {
#pragma unroll
        for (int r = 1; r < 8; ++r) {  // class 0 contributes zero loss; skip
            atomicAdd(&smem[r],      (unsigned int)accP[r]);
            atomicAdd(&smem[20 + r], (unsigned int)accL[r]);
            atomicAdd(&smem[40 + r], (unsigned int)accM[r]);
        }
    }
    if (lane == 16) {
#pragma unroll
        for (int r = 0; r < 8; ++r) {
            atomicAdd(&smem[8 + r],  (unsigned int)accP[r]);
            atomicAdd(&smem[28 + r], (unsigned int)accL[r]);
            atomicAdd(&smem[48 + r], (unsigned int)accM[r]);
        }
    }
#pragma unroll
    for (int c = 0; c < 4; ++c) {
        unsigned int sp = bytesum(hiP[c]);
        unsigned int sl = bytesum(hiL[c]);
        unsigned int sm = bytesum(hiM[c]);
#pragma unroll
        for (int off = 16; off > 0; off >>= 1) {
            sp += (unsigned int)__shfl_xor((int)sp, off);
            sl += (unsigned int)__shfl_xor((int)sl, off);
            sm += (unsigned int)__shfl_xor((int)sm, off);
        }
        if (lane == 0) {
            atomicAdd(&smem[16 + c], sp);
            atomicAdd(&smem[36 + c], sl);
            atomicAdd(&smem[56 + c], sm);
        }
    }
    __syncthreads();
    if (threadIdx.x < 60) atomicAdd(&bins[threadIdx.x], smem[threadIdx.x]);
}

__global__ void dice_final(const unsigned int* __restrict__ bins, float* __restrict__ out) {
    if (threadIdx.x == 0 && blockIdx.x == 0) {
        float s = 0.0f;  // class 0 contributes exactly 0 (binarized tensors are all-zero)
        for (int c = 1; c < NUM_CLASSES; ++c) {
            float inter = (float)bins[40 + c];
            float uni   = (float)bins[c] + (float)bins[20 + c];
            float d = 1.0f - (2.0f * inter + 1e-6f) / (uni + 1e-6f);
            d = fminf(fmaxf(d, 0.0f), 1.0f);
            s += d;
        }
        out[0] = s / (float)NUM_CLASSES;
    }
}

extern "C" void kernel_launch(void* const* d_in, const int* in_sizes, int n_in,
                              void* d_out, int out_size, void* d_ws, size_t ws_size,
                              hipStream_t stream) {
    const float* preds  = (const float*)d_in[0];
    const float* labels = (const float*)d_in[1];
    unsigned int* bins  = (unsigned int*)d_ws;   // needs 64 * 4 = 256 bytes
    float* out          = (float*)d_out;
    const int n = in_sizes[0];

    int blocks = (n + 1023) / 1024;              // 1024 elements per block-trip
    if (blocks > 16384) blocks = 16384;          // grid-stride beyond; keeps SWAR bytes tiny
    if (blocks < 1) blocks = 1;

    hipLaunchKernelGGL(dice_zero_bins, dim3(1), dim3(64), 0, stream, bins);
    hipLaunchKernelGGL(dice_hist, dim3(blocks), dim3(256), 0, stream, preds, labels, bins, n);
    hipLaunchKernelGGL(dice_final, dim3(1), dim3(32), 0, stream, bins, out);
}
